// RinnaClassifierSpan2DecodeIntegrated_74749610820201
// MI455X (gfx1250) — compile-verified, matrix operand order checked by reading, not run on hardware
//
#include <hip/hip_runtime.h>
#include <hip/hip_bf16.h>

// ---------------------------------------------------------------------------
// Problem constants (from the reference)
// ---------------------------------------------------------------------------
#define BATCH 4
#define LTOK  32
#define HDIM  2816            // K dimension, = 88 * 32
#define ODIM  64
#define PAIRS (LTOK * LTOK)   // 1024
#define W1COLS (2 * HDIM + 1) // 5633 (odd stride in the source weight)
#define NGEMM1 (2 * HDIM)     // 5632

#define PANEL_ELEMS (16 * HDIM)        // 45056 bf16 elements per A panel
#define PANEL_BYTES (PANEL_ELEMS * 2)  // 90112 bytes of dynamic LDS
#define PANEL_VECS  (PANEL_BYTES / 16) // 5632 x 16B chunks

typedef __attribute__((ext_vector_type(16))) __bf16 v16bf;
typedef __attribute__((ext_vector_type(8)))  float  v8f;

union Frag16 {
    v16bf v;
    unsigned short s[16];
    uint4 q[2];
};

__device__ __forceinline__ unsigned short f2bf(float f) {
    unsigned u = __builtin_bit_cast(unsigned, f);
    u += 0x7FFFu + ((u >> 16) & 1u);          // round-to-nearest-even bf16
    return (unsigned short)(u >> 16);
}

// ---------------------------------------------------------------------------
// gfx1250 async global->LDS staging (ASYNCcnt path), with portable fallback.
// The builtin's params are int-vector(16B) pointers in AS1 (global) / AS3 (LDS).
// ---------------------------------------------------------------------------
#if defined(__has_builtin)
#if __has_builtin(__builtin_amdgcn_global_load_async_to_lds_b128)
#define HAVE_ASYNC_LDS 1
#endif
#endif

#if defined(HAVE_ASYNC_LDS)
typedef int v4i_ __attribute__((vector_size(16)));
typedef __attribute__((address_space(1))) v4i_* gv4i_t;
typedef __attribute__((address_space(3))) v4i_* lv4i_t;
#endif

// Copy one contiguous 90112-byte panel (16 rows x 2816 bf16) into LDS.
// block = 128 threads; each thread moves 44 x 16B chunks.
__device__ __forceinline__ void stage_panel(const unsigned short* __restrict__ gsrc,
                                            unsigned short* lds) {
    const int t = threadIdx.x;
#if defined(HAVE_ASYNC_LDS)
    gv4i_t g = (gv4i_t)(unsigned long long)gsrc;
    lv4i_t l = (lv4i_t)(unsigned int)(unsigned long long)lds;
#pragma unroll 4
    for (int i = 0; i < PANEL_VECS / 128; ++i) {
        int idx = t + i * 128;
        __builtin_amdgcn_global_load_async_to_lds_b128(g + idx, l + idx, 0, 0);
    }
    asm volatile("s_wait_asynccnt 0" ::: "memory");
#else
    const uint4* g = (const uint4*)gsrc;
    uint4* l = (uint4*)lds;
#pragma unroll 4
    for (int i = 0; i < PANEL_VECS / 128; ++i) {
        int idx = t + i * 128;
        l[idx] = g[idx];
    }
#endif
    __syncthreads();
}

// ---------------------------------------------------------------------------
// One-shot f32 -> bf16 conversion of all WMMA operands into packed layouts:
//   rows [0, 5632)        : W1b[n][k]  (n < 2816 -> W1[:, :H], else W1[:, H:2H])
//   rows [5632, 5696)     : W2b[o][k]
//   rows [5696, 5824)     : vecsb[m][k] = hidden_states[b, 1+t, k]
// grid = 5824 blocks x 256 threads, 11 iters cover K = 2816.
// ---------------------------------------------------------------------------
__global__ void cvt_k(const float* __restrict__ hs,
                      const float* __restrict__ W1,
                      const float* __restrict__ W2,
                      unsigned short* __restrict__ W1b,
                      unsigned short* __restrict__ W2b,
                      unsigned short* __restrict__ vecsb) {
    const int r = blockIdx.x;
    const float* src;
    unsigned short* dst;
    if (r < NGEMM1) {
        src = (r < HDIM) ? (W1 + (size_t)r * W1COLS)
                         : (W1 + (size_t)(r - HDIM) * W1COLS + HDIM);
        dst = W1b + (size_t)r * HDIM;
    } else if (r < NGEMM1 + ODIM) {
        const int o = r - NGEMM1;
        src = W2 + (size_t)o * HDIM;
        dst = W2b + (size_t)o * HDIM;
    } else {
        const int m = r - NGEMM1 - ODIM;
        src = hs + (size_t)((m >> 5) * (LTOK + 1) + 1 + (m & 31)) * HDIM;
        dst = vecsb + (size_t)m * HDIM;
    }
#pragma unroll
    for (int cT = 0; cT < 11; ++cT) {
        int k = threadIdx.x + cT * 256;
        dst[k] = f2bf(src[k]);
    }
}

// Extract last column of W1 (the "ind" weight vector), strided gather -> dense
__global__ void w1c_k(const float* __restrict__ W1, float* __restrict__ w1c) {
    int k = blockIdx.x * 256 + threadIdx.x;
    if (k < HDIM) w1c[k] = W1[(size_t)k * W1COLS + 2 * HDIM];
}

// ---------------------------------------------------------------------------
// GEMM1: AB[m, n] = sum_k vecs[m, k] * W1half[n, k]
//   M = 128, N = 5632, K = 2816.  A panel staged to LDS (async), B from bf16 ws.
// grid = (88, 8), block = 128 (4 waves; wave w -> ntile = bx*4 + w), 88KB LDS
// ---------------------------------------------------------------------------
__global__ void gemm1_k(const unsigned short* __restrict__ vecsb,
                        const unsigned short* __restrict__ W1b,
                        float* __restrict__ AB) {
    extern __shared__ unsigned short apanel[];
    const int mtile = blockIdx.y;                // 0..7
    stage_panel(vecsb + (size_t)mtile * PANEL_ELEMS, apanel);

    const int lane = threadIdx.x & 31;
    const int wave = threadIdx.x >> 5;
    const int n0   = (blockIdx.x * 4 + wave) * 16;   // 0..5616
    const int kb   = (lane >> 4) * 8;                // per-lane K sub-base
    const unsigned short* arow = apanel + (lane & 15) * HDIM;            // LDS
    const unsigned short* brow = W1b + (size_t)(n0 + (lane & 15)) * HDIM;

    v8f c = {0.f, 0.f, 0.f, 0.f, 0.f, 0.f, 0.f, 0.f};

    for (int kk = 0; kk < HDIM; kk += 64) {
        if (kk + 64 < HDIM)
            __builtin_prefetch(brow + kk + 64 + kb, 0, 1);   // global_prefetch_b8
        Frag16 a0, a1, b0, b1;
        a0.q[0] = *(const uint4*)(arow + kk + kb);           // ds_load_b128
        a0.q[1] = *(const uint4*)(arow + kk + kb + 16);
        a1.q[0] = *(const uint4*)(arow + kk + 32 + kb);
        a1.q[1] = *(const uint4*)(arow + kk + 32 + kb + 16);
        b0.q[0] = *(const uint4*)(brow + kk + kb);           // global_load_b128
        b0.q[1] = *(const uint4*)(brow + kk + kb + 16);
        b1.q[0] = *(const uint4*)(brow + kk + 32 + kb);
        b1.q[1] = *(const uint4*)(brow + kk + 32 + kb + 16);
        c = __builtin_amdgcn_wmma_f32_16x16x32_bf16(false, a0.v, false, b0.v,
                                                    (short)0, c, false, false);
        c = __builtin_amdgcn_wmma_f32_16x16x32_bf16(false, a1.v, false, b1.v,
                                                    (short)0, c, false, false);
    }

    const int mloc = (lane >> 4) * 8;
    const int nloc = lane & 15;
    float* dst = AB + (size_t)(mtile * 16) * NGEMM1 + n0 + nloc;
#pragma unroll
    for (int r = 0; r < 8; ++r)
        dst[(size_t)(mloc + r) * NGEMM1] = c[r];
}

// ---------------------------------------------------------------------------
// h-build: hmat[p', k] = bf16( relu( A[b,i,k] + B[b,j,k] + ind*w1c[k] + b1[k] ) )
// grid = 4096 blocks (one per pair-row), block = 256
// ---------------------------------------------------------------------------
__global__ void hbuild_k(const float* __restrict__ AB,
                         const float* __restrict__ w1c,
                         const float* __restrict__ b1,
                         const int*   __restrict__ pred,
                         unsigned short* __restrict__ hmat) {
    const int pidx = blockIdx.x;        // b*1024 + i*32 + j
    const int b    = pidx >> 10;
    const int rem  = pidx & 1023;
    const int i    = rem >> 5;
    const int j    = rem & 31;
    const int s    = pred[b * 2 + 0];
    const int e    = pred[b * 2 + 1];
    float ind = 0.0f;
    if (i == s && j == e)                ind = 2.0f;
    else if (i >= s && j <= e && i <= j) ind = 1.0f;

    const float* ra = AB + (size_t)(b * LTOK + i) * NGEMM1;          // W1a part
    const float* rb = AB + (size_t)(b * LTOK + j) * NGEMM1 + HDIM;   // W1b part
    unsigned short* dst = hmat + (size_t)pidx * HDIM;

#pragma unroll
    for (int cT = 0; cT < 11; ++cT) {
        int k = threadIdx.x + cT * 256;
        float h = ra[k] + rb[k] + ind * w1c[k] + b1[k];
        dst[k] = f2bf(fmaxf(h, 0.0f));
    }
}

// ---------------------------------------------------------------------------
// GEMM2: opre[m, o] = sum_k hmat[m, k] * W2b[o, k]   (M=4096, N=64, K=2816)
// grid = 256 (M tiles), block = 128 (wave w -> o-tile w), A panel via LDS async
// ---------------------------------------------------------------------------
__global__ void gemm2_k(const unsigned short* __restrict__ hmat,
                        const unsigned short* __restrict__ W2b,
                        float* __restrict__ opre) {
    extern __shared__ unsigned short apanel[];
    const int mtile = blockIdx.x;                // 0..255
    stage_panel(hmat + (size_t)mtile * PANEL_ELEMS, apanel);

    const int lane = threadIdx.x & 31;
    const int wave = threadIdx.x >> 5;           // o-tile 0..3
    const int o0   = wave * 16;
    const int kb   = (lane >> 4) * 8;
    const unsigned short* arow = apanel + (lane & 15) * HDIM;
    const unsigned short* brow = W2b + (size_t)(o0 + (lane & 15)) * HDIM;

    v8f c = {0.f, 0.f, 0.f, 0.f, 0.f, 0.f, 0.f, 0.f};

    for (int kk = 0; kk < HDIM; kk += 64) {
        if (kk + 64 < HDIM)
            __builtin_prefetch(brow + kk + 64 + kb, 0, 1);
        Frag16 a0, a1, b0, b1;
        a0.q[0] = *(const uint4*)(arow + kk + kb);
        a0.q[1] = *(const uint4*)(arow + kk + kb + 16);
        a1.q[0] = *(const uint4*)(arow + kk + 32 + kb);
        a1.q[1] = *(const uint4*)(arow + kk + 32 + kb + 16);
        b0.q[0] = *(const uint4*)(brow + kk + kb);
        b0.q[1] = *(const uint4*)(brow + kk + kb + 16);
        b1.q[0] = *(const uint4*)(brow + kk + 32 + kb);
        b1.q[1] = *(const uint4*)(brow + kk + 32 + kb + 16);
        c = __builtin_amdgcn_wmma_f32_16x16x32_bf16(false, a0.v, false, b0.v,
                                                    (short)0, c, false, false);
        c = __builtin_amdgcn_wmma_f32_16x16x32_bf16(false, a1.v, false, b1.v,
                                                    (short)0, c, false, false);
    }

    const int mloc = (lane >> 4) * 8;
    const int nloc = lane & 15;
    float* dst = opre + (size_t)(mtile * 16) * ODIM + o0 + nloc;
#pragma unroll
    for (int r = 0; r < 8; ++r)
        dst[(size_t)(mloc + r) * ODIM] = c[r];
}

// ---------------------------------------------------------------------------
// Epilogue: bias + mask, log_softmax over the P axis, transposed store.
// grid = B*OUT = 256 blocks; block = 256 threads; P = 1024 (4 per thread)
// ---------------------------------------------------------------------------
__global__ void lsm_k(const float* __restrict__ opre,
                      const float* __restrict__ b2,
                      const int*   __restrict__ avail,
                      float* __restrict__ out) {
    __shared__ float red[256];
    const int bo = blockIdx.x;
    const int b  = bo >> 6;
    const int o  = bo & 63;
    const int t  = threadIdx.x;
    const float bias = b2[o];

    float v[4];
    float mx = -3.402823466e38f;
#pragma unroll
    for (int q = 0; q < 4; ++q) {
        int p = t + q * 256;
        float val = (avail[p] >= 1)
                  ? (opre[(size_t)(b * PAIRS + p) * ODIM + o] + bias) : 0.0f;
        v[q] = val;
        mx = fmaxf(mx, val);
    }
    red[t] = mx;  __syncthreads();
    for (int s = 128; s > 0; s >>= 1) {
        if (t < s) red[t] = fmaxf(red[t], red[t + s]);
        __syncthreads();
    }
    mx = red[0];  __syncthreads();

    float sum = 0.0f;
#pragma unroll
    for (int q = 0; q < 4; ++q) sum += __expf(v[q] - mx);
    red[t] = sum; __syncthreads();
    for (int s = 128; s > 0; s >>= 1) {
        if (t < s) red[t] += red[t + s];
        __syncthreads();
    }
    const float lse = mx + __logf(red[0]);

#pragma unroll
    for (int q = 0; q < 4; ++q) {
        int p = t + q * 256;
        out[((size_t)(b * ODIM + o)) * PAIRS + p] = v[q] - lse;
    }
}

// ---------------------------------------------------------------------------
// Launch
// ---------------------------------------------------------------------------
extern "C" void kernel_launch(void* const* d_in, const int* in_sizes, int n_in,
                              void* d_out, int out_size, void* d_ws, size_t ws_size,
                              hipStream_t stream) {
    const float* hs    = (const float*)d_in[0];   // (4, 33, 2816) f32
    const int*   pred  = (const int*)  d_in[1];   // (4, 2) i32
    const int*   avail = (const int*)  d_in[2];   // (32, 32) i32
    const float* W1    = (const float*)d_in[3];   // (2816, 5633) f32
    const float* b1    = (const float*)d_in[4];   // (2816,) f32
    const float* W2    = (const float*)d_in[5];   // (64, 2816) f32
    const float* b2    = (const float*)d_in[6];   // (64,) f32
    float* out = (float*)d_out;                   // (4, 64, 1024) f32

    // Workspace layout (16B aligned offsets), ~60 MB total:
    char* ws = (char*)d_ws;
    float*          AB    = (float*)(ws);                    //  2,883,584 B
    float*          w1c   = (float*)(ws + 2883584);          //     11,264 B
    float*          opre  = (float*)(ws + 2894848);          //  1,048,576 B
    unsigned short* hmat  = (unsigned short*)(ws + 3943424); // 23,068,672 B
    unsigned short* vecsb = (unsigned short*)(ws + 27012096);//    720,896 B
    unsigned short* W1b   = (unsigned short*)(ws + 27732992);// 31,719,424 B
    unsigned short* W2b   = (unsigned short*)(ws + 59452416);//    360,448 B -> 59,812,864

    cvt_k  <<<NGEMM1 + ODIM + BATCH * LTOK, 256, 0, stream>>>(hs, W1, W2, W1b, W2b, vecsb);
    w1c_k  <<<11, 256, 0, stream>>>(W1, w1c);
    gemm1_k<<<dim3(88, 8), 128, PANEL_BYTES, stream>>>(vecsb, W1b, AB);
    hbuild_k<<<BATCH * PAIRS, 256, 0, stream>>>(AB, w1c, b1, pred, hmat);
    gemm2_k<<<256, 128, PANEL_BYTES, stream>>>(hmat, W2b, opre);
    lsm_k  <<<BATCH * ODIM, 256, 0, stream>>>(opre, b2, avail, out);
}
